// DeformableConv2d_16398185136658
// MI455X (gfx1250) — matrix operand order repaired and checked by reading, not compiled, not run on hardware
//
#include <hip/hip_runtime.h>
#include <cstdint>

typedef __attribute__((ext_vector_type(16))) _Float16 v16h;
typedef __attribute__((ext_vector_type(8)))  float    v8f;

#define BATCH   8
#define CIN_    128
#define COUT_   128
#define HH      64
#define WW      64
#define TAPS    9
#define NPIX    4096                 // per image (64*64)
#define NPIXTOT (BATCH * NPIX)       // 32768
#define KDIM    (CIN_ * TAPS)        // 1152
#define KCH     32                   // K chunk per WMMA pass
#define TILE_N  128                  // pixels per workgroup
#define EPSV    1e-5f

// ---------------------------------------------------------------------------
// Kernel 1: offset conv — 18 channels, 3x3, pad 1. One thread per pixel,
// 18 register accumulators; weight loads are loop-uniform -> scalar loads.
// ---------------------------------------------------------------------------
__global__ __launch_bounds__(256) void offset_conv_kernel(
    const float* __restrict__ x, const float* __restrict__ w_off,
    float* __restrict__ off_out) {
  int p  = blockIdx.x * blockDim.x + threadIdx.x;   // 0..32767
  int b  = p >> 12, q = p & 4095;
  int ho = q >> 6,  wo = q & 63;
  float acc[18];
#pragma unroll
  for (int o = 0; o < 18; ++o) acc[o] = 0.f;
  const float* xb = x + ((size_t)b << 7 << 12);     // b*128*4096
  for (int c = 0; c < CIN_; ++c) {
    const float* xp = xb + ((size_t)c << 12);
    const float* wp = w_off + c * TAPS;
#pragma unroll
    for (int ky = 0; ky < 3; ++ky) {
      int y = ho + ky - 1;
      if ((unsigned)y >= 64u) continue;
#pragma unroll
      for (int kx = 0; kx < 3; ++kx) {
        int xx = wo + kx - 1;
        if ((unsigned)xx >= 64u) continue;
        float xv = xp[y * 64 + xx];
        int wi = ky * 3 + kx;
#pragma unroll
        for (int o = 0; o < 18; ++o)
          acc[o] = fmaf(xv, wp[(size_t)o * KDIM + wi], acc[o]);
      }
    }
  }
#pragma unroll
  for (int o = 0; o < 18; ++o)
    off_out[(((size_t)b * 18 + o) << 12) + q] = acc[o];
}

// ---------------------------------------------------------------------------
// Kernel 2: bilinear coordinate table. Per (b, tap, pixel): 4 clamped flat
// indices + 4 validity-masked weights. Channel-independent => computed once.
// ---------------------------------------------------------------------------
__global__ __launch_bounds__(256) void coords_kernel(
    const float* __restrict__ off, int4* __restrict__ cidx,
    float4* __restrict__ cw) {
  int e  = blockIdx.x * blockDim.x + threadIdx.x;   // [0, B*9*4096)
  int q  = e & 4095;
  int t  = (e >> 12) % TAPS;
  int b  = e / (TAPS * NPIX);
  int ho = q >> 6, wo = q & 63;
  float dy = off[(((size_t)b * 18 + 2 * t)     << 12) + q];
  float dx = off[(((size_t)b * 18 + 2 * t + 1) << 12) + q];
  float ys = (float)(ho - 1 + t / 3) + dy;
  float xs = (float)(wo - 1 + t % 3) + dx;
  // clamp before int conversion (offsets are tiny; this only guards UB)
  float y0f = floorf(fminf(fmaxf(ys, -1024.f), 1024.f));
  float x0f = floorf(fminf(fmaxf(xs, -1024.f), 1024.f));
  float wy = ys - y0f, wx = xs - x0f;
  int iy0 = (int)y0f, ix0 = (int)x0f;
  int iy1 = iy0 + 1,  ix1 = ix0 + 1;
  float vy0 = ((unsigned)iy0 < 64u) ? 1.f : 0.f;
  float vy1 = ((unsigned)iy1 < 64u) ? 1.f : 0.f;
  float vx0 = ((unsigned)ix0 < 64u) ? 1.f : 0.f;
  float vx1 = ((unsigned)ix1 < 64u) ? 1.f : 0.f;
  int cy0 = min(max(iy0, 0), 63), cy1 = min(max(iy1, 0), 63);
  int cx0 = min(max(ix0, 0), 63), cx1 = min(max(ix1, 0), 63);
  int4 id;
  id.x = cy0 * 64 + cx0;  id.y = cy0 * 64 + cx1;
  id.z = cy1 * 64 + cx0;  id.w = cy1 * 64 + cx1;
  float4 w;
  w.x = (1.f - wy) * (1.f - wx) * vy0 * vx0;
  w.y = (1.f - wy) * wx         * vy0 * vx1;
  w.z = wy         * (1.f - wx) * vy1 * vx0;
  w.w = wy         * wx         * vy1 * vx1;
  cidx[e] = id;
  cw[e]   = w;
}

// Kernel 3: convert deformable weights to f16 once (k index = c*9 + tap).
__global__ void wconv_kernel(const float* __restrict__ w,
                             _Float16* __restrict__ wh, int n) {
  int i = blockIdx.x * blockDim.x + threadIdx.x;
  if (i < n) wh[i] = (_Float16)w[i];
}

// Kernel 4: zero BN statistics (sum / sumsq, 2*COUT floats).
__global__ void zero_stats_kernel(float* __restrict__ s) {
  s[threadIdx.x] = 0.f;
}

// ---------------------------------------------------------------------------
// Kernel 5: fused gather + WMMA GEMM + bias + BN-stats.
// WG = 256 threads = 8 waves. Tile: 128 channels x 128 pixels.
// Wave w owns channels [16w, 16w+16); K loop in chunks of 32 (f16 WMMA).
// ---------------------------------------------------------------------------
__global__ __launch_bounds__(256) void deform_gemm_kernel(
    const float* __restrict__ x, const _Float16* __restrict__ wh,
    const int4* __restrict__ cidx, const float4* __restrict__ cw,
    const float* __restrict__ b_def, float* __restrict__ out,
    float* __restrict__ stats) {
  __shared__ _Float16 ldsB[TILE_N * KCH];   // [pixel][k], 8 KB

  const int tid   = threadIdx.x;
  const int lane  = tid & 31;
  const int wave  = tid >> 5;
  const int ptile = blockIdx.x * TILE_N;    // global pixel base
  const int b     = ptile >> 12;            // image (tile never crosses images)
  const int qbase = ptile & 4095;
  const int mbase = wave * 16;
  const int row   = lane & 15;
  const int half  = lane >> 4;
  const float* xb = x + ((size_t)b << 7 << 12);

  v8f acc[8];
#pragma unroll
  for (int j = 0; j < 8; ++j)
#pragma unroll
    for (int i = 0; i < 8; ++i) acc[j][i] = 0.f;

  for (int kb = 0; kb < KDIM; kb += KCH) {
    // ---- gather phase: 32 k-values x 128 pixels into LDS (f16) ----
    for (int it = 0; it < 16; ++it) {
      int elem = it * 256 + tid;            // wave handles one pixel, lanes = k
      int pixl = elem >> 5;
      int kl   = elem & 31;
      int kf   = kb + kl;
      int c    = kf / TAPS;
      int t    = kf - c * TAPS;
      int e    = ((b * TAPS + t) << 12) + qbase + pixl;
      int4   id = cidx[e];
      float4 w  = cw[e];
      const float* xp = xb + ((size_t)c << 12);
      float v = w.x * xp[id.x] + w.y * xp[id.y] +
                w.z * xp[id.z] + w.w * xp[id.w];
      ldsB[pixl * KCH + kl] = (_Float16)v;
    }
    __syncthreads();

    // ---- A fragment: weights, ISA 16-bit A 16x32 layout ----
    // lane<16: K = {0,1,..,7} & {16..23}; lane>=16: +8 within each 16-group.
    union { v16h v; uint32_t u[8]; } afrag;
    const _Float16* wrow = wh + (size_t)(mbase + row) * KDIM + kb + half * 8;
#pragma unroll
    for (int v4 = 0; v4 < 4; ++v4)
      afrag.u[v4]     = *(const uint32_t*)(wrow + 2 * v4);
#pragma unroll
    for (int v4 = 0; v4 < 4; ++v4)
      afrag.u[4 + v4] = *(const uint32_t*)(wrow + 16 + 2 * v4);

    // ---- 8 N-tiles of 16 pixels each ----
#pragma unroll
    for (int j = 0; j < 8; ++j) {
      // B fragment: N = lane%16, K = half*8 + {0..7} & 16 + half*8 + {0..7}
      union { v16h v; uint32_t u[8]; } bfrag;
      const _Float16* bp = &ldsB[(j * 16 + row) * KCH + half * 8];
#pragma unroll
      for (int v4 = 0; v4 < 8; ++v4)
        bfrag.u[v4] = *(const uint32_t*)(bp + 2 * (v4 & 3) + (v4 >> 2) * 16);
      acc[j] = __builtin_amdgcn_wmma_f32_16x16x32_f16(
          false, afrag.v, false, bfrag.v, (short)0, acc[j], false, false);
    }
    __syncthreads();
  }

  // ---- epilogue: bias, store, fused BN statistics ----
#pragma unroll
  for (int r = 0; r < 8; ++r) {
    int ch = mbase + r + half * 8;          // C/D layout: M = r + 8*(lane>=16)
    float bias = b_def[ch];
    float s0 = 0.f, s1 = 0.f;
    float* op = out + ((((size_t)b << 7) + ch) << 12) + qbase + row;
#pragma unroll
    for (int j = 0; j < 8; ++j) {
      float v = acc[j][r] + bias;
      op[j * 16] = v;
      s0 += v;
      s1 += v * v;
    }
    // reduce across the 16 lanes sharing this channel (N dimension)
#pragma unroll
    for (int m = 1; m < 16; m <<= 1) {
      s0 += __shfl_xor(s0, m, 32);
      s1 += __shfl_xor(s1, m, 32);
    }
    if (row == 0) {
      atomicAdd(&stats[ch],         s0);
      atomicAdd(&stats[COUT_ + ch], s1);
    }
  }
}

// ---------------------------------------------------------------------------
// Kernel 6: batchnorm (population stats) + SiLU, in place on d_out.
// ---------------------------------------------------------------------------
__global__ __launch_bounds__(256) void bn_silu_kernel(
    float* __restrict__ out, const float* __restrict__ stats,
    const float* __restrict__ gamma, const float* __restrict__ beta) {
  int i  = blockIdx.x * blockDim.x + threadIdx.x;
  int ch = (i >> 12) & 127;
  const float invN = 1.0f / (float)NPIXTOT;
  float mean = stats[ch] * invN;
  float var  = stats[COUT_ + ch] * invN - mean * mean;
  float scale = gamma[ch] * rsqrtf(var + EPSV);
  float v = (out[i] - mean) * scale + beta[ch];
  out[i] = v / (1.0f + __expf(-v));          // v * sigmoid(v)
}

// ---------------------------------------------------------------------------
extern "C" void kernel_launch(void* const* d_in, const int* in_sizes, int n_in,
                              void* d_out, int out_size, void* d_ws,
                              size_t ws_size, hipStream_t stream) {
  const float* x     = (const float*)d_in[0];
  const float* w_off = (const float*)d_in[1];
  const float* w_def = (const float*)d_in[2];
  const float* b_def = (const float*)d_in[3];
  const float* gamma = (const float*)d_in[4];
  const float* beta  = (const float*)d_in[5];
  float* out = (float*)d_out;

  // workspace carving (≈12.1 MB total, 16B aligned slabs)
  char* ws = (char*)d_ws;
  size_t o0 = 0;
  float*    ws_off  = (float*)(ws + o0);  o0 += (size_t)BATCH * 18 * NPIX * 4;
  int4*     ws_idx  = (int4*)(ws + o0);   o0 += (size_t)BATCH * TAPS * NPIX * 16;
  float4*   ws_w    = (float4*)(ws + o0); o0 += (size_t)BATCH * TAPS * NPIX * 16;
  _Float16* ws_wh   = (_Float16*)(ws + o0); o0 += (size_t)COUT_ * KDIM * 2;
  float*    ws_stat = (float*)(ws + o0);

  zero_stats_kernel<<<1, 2 * COUT_, 0, stream>>>(ws_stat);
  offset_conv_kernel<<<NPIXTOT / 256, 256, 0, stream>>>(x, w_off, ws_off);
  coords_kernel<<<(BATCH * TAPS * NPIX) / 256, 256, 0, stream>>>(ws_off, ws_idx, ws_w);
  wconv_kernel<<<(COUT_ * KDIM) / 256, 256, 0, stream>>>(w_def, ws_wh, COUT_ * KDIM);
  deform_gemm_kernel<<<NPIXTOT / TILE_N, 256, 0, stream>>>(
      x, ws_wh, ws_idx, ws_w, b_def, out, ws_stat);
  bn_silu_kernel<<<(NPIXTOT * COUT_) / 256, 256, 0, stream>>>(
      out, ws_stat, gamma, beta);
}
